// FaultPropagationGAT_18983755448730
// MI455X (gfx1250) — matrix operand order repaired and checked by reading, not compiled, not run on hardware
//
#include <hip/hip_runtime.h>
#include <hip/hip_bf16.h>

typedef __attribute__((ext_vector_type(16))) _Float16 v16h;
typedef __attribute__((ext_vector_type(8)))  float    v8f;

#define NNODES 100000
#define NEDGES 1600000
#define NHEADS 2
#define NEG_SLOPE 0.2f

// ---------------- helpers ----------------
__device__ __forceinline__ unsigned floatKey(float f) {
    unsigned u = __float_as_uint(f);
    return (u & 0x80000000u) ? ~u : (u | 0x80000000u);
}
__device__ __forceinline__ float keyFloat(unsigned k) {
    return (k & 0x80000000u) ? __uint_as_float(k & 0x7fffffffu) : __uint_as_float(~k);
}
__device__ __forceinline__ float lrelu(float x) { return x > 0.f ? x : NEG_SLOPE * x; }

// ---------------- init ----------------
__global__ void fill_u32(unsigned* __restrict__ p, unsigned v, int n) {
    int t = blockIdx.x * blockDim.x + threadIdx.x;
    if (t < n) p[t] = v;
}

// ---------------- pack W into WMMA B-operand layout (f16) ----------------
// B fragment (16x16x32 f16), k order chosen here to match A: lane L holds column N = L%16, K = (L<16?0:16)+j, j=0..15.
// Storage: halves at [((nt*KT+kt)*32 + lane)*16 + j]  -> one coalesced v16h (32B) load per lane.
__global__ void pack_b_single(const float* __restrict__ W, unsigned short* __restrict__ Bp) {
    int t = blockIdx.x * blockDim.x + threadIdx.x;   // K=128 (KT=4), N=128 (NT=8)
    if (t >= 4 * 8 * 512) return;
    int j    = t & 15;
    int lane = (t >> 4) & 31;
    int frag = t >> 9;
    int kt = frag & 3;
    int nt = frag >> 2;
    int n = nt * 16 + (lane & 15);
    int k = kt * 32 + ((lane >> 4) << 4) + j;
    _Float16 h = (_Float16)W[k * 128 + n];
    Bp[t] = *(const unsigned short*)&h;
}

__global__ void pack_b_dual(const float* __restrict__ W2, const float* __restrict__ resW2,
                            unsigned short* __restrict__ Bp) {
    // combined 128x128: cols 0..63 = W2 (128x64), cols 64..127 = resW2 (128x64)
    int t = blockIdx.x * blockDim.x + threadIdx.x;
    if (t >= 4 * 8 * 512) return;
    int j    = t & 15;
    int lane = (t >> 4) & 31;
    int frag = t >> 9;
    int kt = frag & 3;
    int nt = frag >> 2;
    int n = nt * 16 + (lane & 15);
    int k = kt * 32 + ((lane >> 4) << 4) + j;
    float v = (n < 64) ? W2[k * 64 + n] : resW2[k * 64 + (n - 64)];
    _Float16 h = (_Float16)v;
    Bp[t] = *(const unsigned short*)&h;
}

// ---------------- GEMM: C[100000x128] = A[100000x128] @ Bpack(128x128) ----------------
// One wave per 16 rows; 8 accumulator tiles cover all 128 cols; K-loop of 4 WMMAs per tile.
__global__ void gemm_wmma_f16(const float* __restrict__ A,
                              const unsigned short* __restrict__ Bpack,
                              float* __restrict__ C, int Mtiles) {
    constexpr int KT = 4;
    constexpr int NT = 8;
    int lane = threadIdx.x & 31;
    int mt   = blockIdx.x * (blockDim.x >> 5) + (threadIdx.x >> 5);
    if (mt >= Mtiles) return;                 // wave-uniform: EXEC all-ones at WMMA
    int halfsel = lane >> 4;                  // 0 or 1
    int row     = mt * 16 + (lane & 15);
    const float* arow = A + (size_t)row * 128;

    v8f zero = {};
    v8f c[NT];
#pragma unroll
    for (int i = 0; i < NT; ++i) c[i] = zero;

    const v16h* bbase = (const v16h*)Bpack;
#pragma unroll
    for (int kt = 0; kt < KT; ++kt) {
        // A frag: K = kt*32 + halfsel*16 + j, j=0..15
        const float* p0 = arow + kt * 32 + halfsel * 16;
        const float* p1 = p0 + 8;
        v16h a;
#pragma unroll
        for (int j = 0; j < 8; ++j) {
            a[j]     = (_Float16)p0[j];
            a[j + 8] = (_Float16)p1[j];
        }
#pragma unroll
        for (int nt = 0; nt < NT; ++nt) {
            v16h b = bbase[(nt * KT + kt) * 32 + lane];
            c[nt] = __builtin_amdgcn_wmma_f32_16x16x32_f16(
                false, a, false, b, (short)0, c[nt], false, false);
        }
    }
    // C layout: VGPR r -> M = mt*16 + halfsel*8 + r, N = nt*16 + lane%16
    int mrow = mt * 16 + halfsel * 8;
    int ncol = lane & 15;
#pragma unroll
    for (int nt = 0; nt < NT; ++nt) {
        float* crow = C + (size_t)mrow * 128 + nt * 16 + ncol;
#pragma unroll
        for (int r = 0; r < 8; ++r) crow[(size_t)r * 128] = c[nt][r];
    }
}

// ---------------- attention scores: one wave per (node,head), dout=64 ----------------
__global__ void attn_scores(const float* __restrict__ feat, const float* __restrict__ al,
                            const float* __restrict__ ar, float* __restrict__ el,
                            float* __restrict__ er, int npairs, int heads) {
    int lane = threadIdx.x & 31;
    int pair = blockIdx.x * (blockDim.x >> 5) + (threadIdx.x >> 5);
    if (pair >= npairs) return;
    int node = pair / heads;
    int h    = pair - node * heads;
    const float* fp  = feat + (size_t)node * 128 + h * 64;
    const float* alp = al + h * 64;
    const float* arp = ar + h * 64;
    float f0 = fp[lane], f1 = fp[lane + 32];
    float ev = f0 * alp[lane] + f1 * alp[lane + 32];
    float rv = f0 * arp[lane] + f1 * arp[lane + 32];
#pragma unroll
    for (int off = 16; off > 0; off >>= 1) {
        ev += __shfl_xor(ev, off, 32);
        rv += __shfl_xor(rv, off, 32);
    }
    if (lane == 0) { el[pair] = ev; er[pair] = rv; }
}

// ---------------- edge softmax passes ----------------
__global__ void edge_max_k(const int* __restrict__ src, const int* __restrict__ dst,
                           const float* __restrict__ el, const float* __restrict__ er,
                           unsigned* __restrict__ mkey, int nE, int heads) {
    int t = blockIdx.x * blockDim.x + threadIdx.x;
    if (t >= nE * heads) return;
    int e = t / heads, h = t - e * heads;
    float x = lrelu(el[src[e] * heads + h] + er[dst[e] * heads + h]);
    atomicMax(mkey + dst[e] * heads + h, floatKey(x));
}

__global__ void edge_exp_k(const int* __restrict__ src, const int* __restrict__ dst,
                           const float* __restrict__ el, const float* __restrict__ er,
                           const unsigned* __restrict__ mkey, float* __restrict__ ex,
                           float* __restrict__ denom, float* __restrict__ deg,
                           int nE, int heads) {
    int t = blockIdx.x * blockDim.x + threadIdx.x;
    if (t >= nE * heads) return;
    int e = t / heads, h = t - e * heads;
    int d = dst[e];
    float x = lrelu(el[src[e] * heads + h] + er[d * heads + h]);
    float m = keyFloat(mkey[d * heads + h]);
    float v = __expf(x - m);
    ex[t] = v;
    atomicAdd(denom + d * heads + h, v);
    if (deg != nullptr && h == 0) atomicAdd(deg + d, 1.0f);
}

// ---------------- message aggregation: agg[dst] += feat[src] * a ----------------
template <int F, int FS, int AS>
__global__ void edge_aggr(const int* __restrict__ src, const int* __restrict__ dst,
                          const float* __restrict__ ex, const float* __restrict__ denom,
                          const float* __restrict__ feat, float* __restrict__ agg,
                          int nE, int heads) {
    constexpr int TPE = F / 4;
    int t = blockIdx.x * blockDim.x + threadIdx.x;
    if (t >= nE * TPE) return;
    int e = t / TPE;
    int q = t - e * TPE;
    int f = q * 4;
    int h = f >> 6;   // dout = 64 per head
    int s = src[e], d = dst[e];
    float a = ex[e * heads + h] / denom[d * heads + h];
    const float4 v = *(const float4*)(feat + (size_t)s * FS + f);
    float* o = agg + (size_t)d * AS + f;
    atomicAdd(o + 0, v.x * a);
    atomicAdd(o + 1, v.y * a);
    atomicAdd(o + 2, v.z * a);
    atomicAdd(o + 3, v.w * a);
}

// ---------------- sigmoid-weighted edge sum ----------------
__global__ void edge_wsum(const int* __restrict__ src, const int* __restrict__ dst,
                          const float* __restrict__ ew, const float* __restrict__ pc,
                          const float* __restrict__ h1, float* __restrict__ s, int nE) {
    constexpr int TPE = 32;  // 128 feats / 4
    int t = blockIdx.x * blockDim.x + threadIdx.x;
    if (t >= nE * TPE) return;
    int e = t / TPE;
    int f = (t - e * TPE) * 4;
    float w = 1.f / (1.f + __expf(-(pc[0] * ew[e])));
    const float4 v = *(const float4*)(h1 + (size_t)src[e] * 128 + f);
    float* o = s + (size_t)dst[e] * 128 + f;
    atomicAdd(o + 0, v.x * w);
    atomicAdd(o + 1, v.y * w);
    atomicAdd(o + 2, v.z * w);
    atomicAdd(o + 3, v.w * w);
}

// ---------------- elementwise epilogues ----------------
__global__ void add_res_bias(float* __restrict__ agg, const float* __restrict__ x,
                             const float* __restrict__ b, int total) {  // h1 = agg + X + b1
    int t = blockIdx.x * blockDim.x + threadIdx.x;
    if (t < total) agg[t] += x[t] + b[t & 127];
}
__global__ void mean_div(float* __restrict__ s, const float* __restrict__ deg, int total) {
    int t = blockIdx.x * blockDim.x + threadIdx.x;
    if (t < total) s[t] = s[t] / fmaxf(deg[t >> 7], 1.0f);
}
__global__ void final_add(float* __restrict__ out, const float* __restrict__ featcat,
                          const float* __restrict__ b2, int total) {  // out += res2 + b2
    int t = blockIdx.x * blockDim.x + threadIdx.x;
    if (t >= total) return;
    int node = t >> 6, c = t & 63;
    out[t] = out[t] + featcat[(size_t)node * 128 + 64 + c] + b2[c];
}

// ---------------- host launch ----------------
static inline int nblk(long n, int b) { return (int)((n + b - 1) / b); }

extern "C" void kernel_launch(void* const* d_in, const int* in_sizes, int n_in,
                              void* d_out, int out_size, void* d_ws, size_t ws_size,
                              hipStream_t stream) {
    (void)in_sizes; (void)n_in; (void)out_size; (void)ws_size;
    const float* X     = (const float*)d_in[0];
    const int*   src   = (const int*)d_in[1];
    const int*   dst   = (const int*)d_in[2];
    const float* ew    = (const float*)d_in[3];
    const float* W1    = (const float*)d_in[4];
    const float* al1   = (const float*)d_in[5];
    const float* ar1   = (const float*)d_in[6];
    const float* b1    = (const float*)d_in[7];
    const float* W2    = (const float*)d_in[8];
    const float* al2   = (const float*)d_in[9];
    const float* ar2   = (const float*)d_in[10];
    const float* resW2 = (const float*)d_in[11];
    const float* b2    = (const float*)d_in[12];
    const float* pc    = (const float*)d_in[13];
    float* out = (float*)d_out;

    char* ws = (char*)d_ws;
    size_t cur = 0;
    auto alloc = [&](size_t bytes) -> char* {
        char* p = ws + cur;
        cur = (cur + bytes + 255) & ~(size_t)255;
        return p;
    };
    float*    feat   = (float*)alloc((size_t)NNODES * 128 * 4);  // feat1, later [feat2|res2]
    float*    agg    = (float*)alloc((size_t)NNODES * 128 * 4);  // agg1 -> h1 (in place)
    float*    smean  = (float*)alloc((size_t)NNODES * 128 * 4);  // weighted sum -> h1w
    float*    ex1    = (float*)alloc((size_t)NEDGES * 2 * 4);
    float*    ex2    = (float*)alloc((size_t)NEDGES * 4);
    float*    el1    = (float*)alloc((size_t)NNODES * 2 * 4);
    float*    er1    = (float*)alloc((size_t)NNODES * 2 * 4);
    unsigned* mkey1  = (unsigned*)alloc((size_t)NNODES * 2 * 4);
    float*    denom1 = (float*)alloc((size_t)NNODES * 2 * 4);
    float*    el2    = (float*)alloc((size_t)NNODES * 4);
    float*    er2    = (float*)alloc((size_t)NNODES * 4);
    unsigned* mkey2  = (unsigned*)alloc((size_t)NNODES * 4);
    float*    denom2 = (float*)alloc((size_t)NNODES * 4);
    float*    deg    = (float*)alloc((size_t)NNODES * 4);
    unsigned short* Bp1 = (unsigned short*)alloc(4 * 8 * 512 * 2);
    unsigned short* Bp2 = (unsigned short*)alloc(4 * 8 * 512 * 2);

    // ---- init ----
    hipMemsetAsync(agg,    0, (size_t)NNODES * 128 * 4, stream);
    hipMemsetAsync(smean,  0, (size_t)NNODES * 128 * 4, stream);
    hipMemsetAsync(denom1, 0, (size_t)NNODES * 2 * 4,   stream);
    hipMemsetAsync(denom2, 0, (size_t)NNODES * 4,       stream);
    hipMemsetAsync(deg,    0, (size_t)NNODES * 4,       stream);
    hipMemsetAsync(out,    0, (size_t)NNODES * 64 * 4,  stream);
    // monotone key of -inf: bits=0xFF800000 (sign set) -> ~bits = 0x007FFFFF
    fill_u32<<<nblk(NNODES * 2, 256), 256, 0, stream>>>(mkey1, 0x007fffffu, NNODES * 2);
    fill_u32<<<nblk(NNODES, 256),     256, 0, stream>>>(mkey2, 0x007fffffu, NNODES);

    // ---- pack weights into WMMA B layout ----
    pack_b_single<<<nblk(16384, 256), 256, 0, stream>>>(W1, Bp1);
    pack_b_dual  <<<nblk(16384, 256), 256, 0, stream>>>(W2, resW2, Bp2);

    // ---- GAT layer 1 ----
    const int MT = NNODES / 16;  // 6250 (exact)
    gemm_wmma_f16<<<nblk(MT, 8), 256, 0, stream>>>(X, Bp1, feat, MT);
    attn_scores<<<nblk((long)NNODES * 2, 8), 256, 0, stream>>>(feat, al1, ar1, el1, er1,
                                                               NNODES * 2, NHEADS);
    edge_max_k<<<nblk((long)NEDGES * 2, 256), 256, 0, stream>>>(src, dst, el1, er1, mkey1,
                                                                NEDGES, NHEADS);
    edge_exp_k<<<nblk((long)NEDGES * 2, 256), 256, 0, stream>>>(src, dst, el1, er1, mkey1,
                                                                ex1, denom1, deg, NEDGES, NHEADS);
    edge_aggr<128, 128, 128><<<nblk((long)NEDGES * 32, 256), 256, 0, stream>>>(
        src, dst, ex1, denom1, feat, agg, NEDGES, NHEADS);
    add_res_bias<<<nblk((long)NNODES * 128, 256), 256, 0, stream>>>(agg, X, b1, NNODES * 128);

    // ---- sigmoid-weighted mean aggregation ----
    edge_wsum<<<nblk((long)NEDGES * 32, 256), 256, 0, stream>>>(src, dst, ew, pc, agg, smean,
                                                                NEDGES);
    mean_div<<<nblk((long)NNODES * 128, 256), 256, 0, stream>>>(smean, deg, NNODES * 128);

    // ---- GAT layer 2 (feat2 and linear residual fused into one 128-col GEMM) ----
    gemm_wmma_f16<<<nblk(MT, 8), 256, 0, stream>>>(smean, Bp2, feat, MT);
    attn_scores<<<nblk((long)NNODES, 8), 256, 0, stream>>>(feat, al2, ar2, el2, er2, NNODES, 1);
    edge_max_k<<<nblk((long)NEDGES, 256), 256, 0, stream>>>(src, dst, el2, er2, mkey2, NEDGES, 1);
    edge_exp_k<<<nblk((long)NEDGES, 256), 256, 0, stream>>>(src, dst, el2, er2, mkey2, ex2,
                                                            denom2, nullptr, NEDGES, 1);
    edge_aggr<64, 128, 64><<<nblk((long)NEDGES * 16, 256), 256, 0, stream>>>(
        src, dst, ex2, denom2, feat, out, NEDGES, 1);
    final_add<<<nblk((long)NNODES * 64, 256), 256, 0, stream>>>(out, feat, b2, NNODES * 64);
}
